// GAT_17489106829715
// MI455X (gfx1250) — matrix operand order repaired and checked by reading, not compiled, not run on hardware
//
#include <hip/hip_runtime.h>
#include <hip/hip_bf16.h>
#include <math.h>

typedef __attribute__((ext_vector_type(16))) _Float16 v16h;
typedef __attribute__((ext_vector_type(8)))  float    v8f;

#define NEG_SLOPE 0.2f
#define NEGINF_ORD 0x007FFFFFu   // ford(-inf)

// ---- order-preserving float<->uint for atomicMax on floats ----
__device__ __forceinline__ unsigned ford(float f) {
  unsigned u = __float_as_uint(f);
  return (u & 0x80000000u) ? ~u : (u | 0x80000000u);
}
__device__ __forceinline__ float unford(unsigned u) {
  return (u & 0x80000000u) ? __uint_as_float(u ^ 0x80000000u)
                           : __uint_as_float(~u);
}

// ---------------- fills ----------------
__global__ void k_fill_f32(float* p, float v, long n) {
  long i = (long)blockIdx.x * blockDim.x + threadIdx.x;
  if (i < n) p[i] = v;
}
__global__ void k_fill_u32(unsigned* p, unsigned v, long n) {
  long i = (long)blockIdx.x * blockDim.x + threadIdx.x;
  if (i < n) p[i] = v;
}

// ---------------- pack W1 [64,512] f32 -> B fragments f16 -------------------
// fragment f = tn*2 + kbIdx covers B rows K = kbIdx*32 .. +31, cols tn*16..+15.
// per-lane layout (g=lane>>4, col=tn*16+(lane&15)): half j holds
//   W1[(kbIdx*32 + g*8 + (j&7) + 16*(j>>3))*512 + col]
// -> GEMM reads one contiguous 32B v16h per lane.
__global__ void k_packB(const float* __restrict__ W1, _Float16* __restrict__ bp) {
  int idx = blockIdx.x * blockDim.x + threadIdx.x;   // 32*2*32*16 = 32768
  if (idx >= 32768) return;
  int j     = idx & 15;
  int lane  = (idx >> 4) & 31;
  int kbIdx = (idx >> 9) & 1;
  int tn    = idx >> 10;
  int g = lane >> 4, m = lane & 15;
  int k = kbIdx * 32 + g * 8 + (j & 7) + ((j >> 3) << 4);
  int col = tn * 16 + m;
  bp[idx] = (_Float16)W1[(long)k * 512 + col];
}

// ---------------- layer-1 GEMM: h1[N,512] = x[N,64] @ W1[64,512] ------------
// one wave per (row-tile, group of 4 column tiles): A loaded once, 8 WMMAs.
__device__ __forceinline__ v16h load_a_frag(const float4* __restrict__ xr,
                                            int kb, int g) {
  // A 16x32 f16 layout: lane m holds K {kb+8g .. +7} and {kb+16+8g .. +7}
  int base = (kb + 8 * g) >> 2;                  // float4 index
  float4 q0 = xr[base],     q1 = xr[base + 1];
  float4 q2 = xr[base + 4], q3 = xr[base + 5];   // +16 floats
  v16h a;
  a[0]=(_Float16)q0.x; a[1]=(_Float16)q0.y; a[2]=(_Float16)q0.z; a[3]=(_Float16)q0.w;
  a[4]=(_Float16)q1.x; a[5]=(_Float16)q1.y; a[6]=(_Float16)q1.z; a[7]=(_Float16)q1.w;
  a[8]=(_Float16)q2.x; a[9]=(_Float16)q2.y; a[10]=(_Float16)q2.z; a[11]=(_Float16)q2.w;
  a[12]=(_Float16)q3.x; a[13]=(_Float16)q3.y; a[14]=(_Float16)q3.z; a[15]=(_Float16)q3.w;
  return a;
}

__global__ void k_gemm1_wmma(const float* __restrict__ x,
                             const _Float16* __restrict__ bp,
                             float* __restrict__ h1, int Nn) {
  int wave = (int)((blockIdx.x * blockDim.x + threadIdx.x) >> 5);
  int lane = threadIdx.x & 31;
  int tilesM = (Nn + 15) >> 4;
  if (wave >= tilesM * 8) return;            // uniform per-wave exit

  int tm  = wave >> 3;                        // row tile
  int tng = wave & 7;                         // group of 4 column tiles
  int g = lane >> 4;
  int m = lane & 15;
  int row = tm * 16 + m;
  int rc = row < Nn ? row : (Nn - 1);         // clamp only: padding rows of A
                                              // affect only unstored D rows
  const float4* xr = (const float4*)(x + (long)rc * 64);
  v16h a0 = load_a_frag(xr, 0, g);
  v16h a1 = load_a_frag(xr, 32, g);

  const v16h* bfr = (const v16h*)bp;
  bool full = (tm * 16 + 16) <= Nn;           // uniform: no divergent stores

#pragma unroll
  for (int t = 0; t < 4; ++t) {
    int tn = tng * 4 + t;
    v16h b0 = bfr[(tn * 2 + 0) * 32 + lane];
    v16h b1 = bfr[(tn * 2 + 1) * 32 + lane];
    v8f acc = {};
    acc = __builtin_amdgcn_wmma_f32_16x16x32_f16(false, a0, false, b0,
                                                 (short)0, acc, false, false);
    acc = __builtin_amdgcn_wmma_f32_16x16x32_f16(false, a1, false, b1,
                                                 (short)0, acc, false, false);
    int col = tn * 16 + m;
    // C/D layout: lanes0-15 -> M=v, lanes16-31 -> M=v+8
    long base = (long)(tm * 16 + 8 * g) * 512 + col;
    if (full) {
#pragma unroll
      for (int v = 0; v < 8; ++v) h1[base + (long)v * 512] = acc[v];
    } else {
#pragma unroll
      for (int v = 0; v < 8; ++v)
        if (tm * 16 + 8 * g + v < Nn) h1[base + (long)v * 512] = acc[v];
    }
  }
}

// ---------------- attention coefficients layer 1: a_s/a_d [N,8] -------------
__global__ void k_att1(const float* __restrict__ h1,
                       const float* __restrict__ att_s,
                       const float* __restrict__ att_d,
                       float* as1, float* ad1, int Nn) {
  long idx = (long)blockIdx.x * blockDim.x + threadIdx.x;
  if (idx >= (long)Nn * 8) return;
  int hd = (int)(idx & 7);
  long n = idx >> 3;
  const float* hp = h1 + n * 512 + hd * 64;
  const float* sp = att_s + hd * 64;
  const float* dp = att_d + hd * 64;
  float s1 = 0.f, s2 = 0.f;
#pragma unroll 8
  for (int c = 0; c < 64; ++c) {
    float hv = hp[c];
    s1 += hv * sp[c];
    s2 += hv * dp[c];
  }
  as1[idx] = s1;
  ad1[idx] = s2;
}

// ---------------- generic edge passes (self-loops appended as edges E..E+N-1) ----
__device__ __forceinline__ void edge_nodes(const long long* ei, long e, int E,
                                           int& s, int& d) {
  if (e < E) { s = (int)ei[e]; d = (int)ei[(long)E + e]; }
  else       { s = d = (int)(e - E); }
}

__global__ void k_edge_max(const long long* __restrict__ ei, int E, int Nn, int H,
                           const float* __restrict__ as, const float* __restrict__ ad,
                           unsigned* __restrict__ mu) {
  long idx = (long)blockIdx.x * blockDim.x + threadIdx.x;
  long tot = (long)(E + Nn) * H;
  if (idx >= tot) return;
  int hd = (int)(idx % H);
  long e = idx / H;
  int s, d; edge_nodes(ei, e, E, s, d);
  float el = as[(long)s * H + hd] + ad[(long)d * H + hd];
  el = el > 0.f ? el : NEG_SLOPE * el;
  atomicMax(&mu[(long)d * H + hd], ford(el));
}

__global__ void k_edge_sum(const long long* __restrict__ ei, int E, int Nn, int H,
                           const float* __restrict__ as, const float* __restrict__ ad,
                           const unsigned* __restrict__ mu, float* __restrict__ den) {
  long idx = (long)blockIdx.x * blockDim.x + threadIdx.x;
  long tot = (long)(E + Nn) * H;
  if (idx >= tot) return;
  int hd = (int)(idx % H);
  long e = idx / H;
  int s, d; edge_nodes(ei, e, E, s, d);
  float el = as[(long)s * H + hd] + ad[(long)d * H + hd];
  el = el > 0.f ? el : NEG_SLOPE * el;
  float mx = unford(mu[(long)d * H + hd]);
  atomicAdd(&den[(long)d * H + hd], __expf(el - mx));
}

__global__ void k_edge_agg(const long long* __restrict__ ei, int E, int Nn, int H, int C,
                           const float* __restrict__ as, const float* __restrict__ ad,
                           const unsigned* __restrict__ mu, const float* __restrict__ den,
                           const float* __restrict__ feat, float* __restrict__ outp) {
  long idx = (long)blockIdx.x * blockDim.x + threadIdx.x;
  long tot = (long)(E + Nn) * H;
  if (idx >= tot) return;
  int hd = (int)(idx % H);
  long e = idx / H;
  int s, d; edge_nodes(ei, e, E, s, d);
  float el = as[(long)s * H + hd] + ad[(long)d * H + hd];
  el = el > 0.f ? el : NEG_SLOPE * el;
  float mx = unford(mu[(long)d * H + hd]);
  float alpha = __expf(el - mx) / den[(long)d * H + hd];
  const float* fp = feat + ((long)s * H + hd) * C;
  float* op = outp + ((long)d * H + hd) * C;
#pragma unroll 4
  for (int c = 0; c < C; ++c)
    atomicAdd(&op[c], alpha * fp[c]);
}

// ---------------- layer-2 GEMV (one wave per node): z=elu(agg1+b1); h2=z@W2 ------
__global__ void k_gemv2(const float* __restrict__ agg1, const float* __restrict__ b1,
                        const float* __restrict__ W2,
                        const float* __restrict__ att_s2, const float* __restrict__ att_d2,
                        float* h2, float* as2, float* ad2, int Nn) {
  int wave = (int)((blockIdx.x * blockDim.x + threadIdx.x) >> 5);
  int lane = threadIdx.x & 31;
  if (wave >= Nn) return;
  float s = 0.f;
  for (int k = lane; k < 512; k += 32) {
    float z = agg1[(long)wave * 512 + k] + b1[k];
    z = z > 0.f ? z : (__expf(z) - 1.f);     // ELU(alpha=1)
    s += z * W2[k];
  }
#pragma unroll
  for (int o = 16; o > 0; o >>= 1) s += __shfl_down(s, o, 32);
  if (lane == 0) {
    h2[wave] = s;
    as2[wave] = s * att_s2[0];
    ad2[wave] = s * att_d2[0];
  }
}

__global__ void k_final(const float* __restrict__ agg2, const float* __restrict__ b2,
                        float* __restrict__ outp, int Nn) {
  long i = (long)blockIdx.x * blockDim.x + threadIdx.x;
  if (i < Nn) outp[i] = agg2[i] + b2[0];
}

// ---------------- host launcher ----------------
static inline int blks(long n, int b) { return (int)((n + b - 1) / b); }

extern "C" void kernel_launch(void* const* d_in, const int* in_sizes, int n_in,
                              void* d_out, int out_size, void* d_ws, size_t ws_size,
                              hipStream_t stream) {
  const float*     x    = (const float*)d_in[0];
  const long long* ei   = (const long long*)d_in[1];
  const float*     W1   = (const float*)d_in[2];
  const float*     atS1 = (const float*)d_in[3];
  const float*     atD1 = (const float*)d_in[4];
  const float*     b1   = (const float*)d_in[5];
  const float*     W2   = (const float*)d_in[6];
  const float*     atS2 = (const float*)d_in[7];
  const float*     atD2 = (const float*)d_in[8];
  const float*     b2   = (const float*)d_in[9];

  const int N = in_sizes[0] / 64;
  const int E = in_sizes[1] / 2;
  const long Etot = (long)E + N;

  // workspace partition (all 4-byte elements; bpack needs 32B alignment)
  float* ws = (float*)d_ws;
  size_t off = 0;
  float*    h1   = ws + off; off += (size_t)N * 512;
  float*    as1  = ws + off; off += (size_t)N * 8;
  float*    ad1  = ws + off; off += (size_t)N * 8;
  unsigned* m1   = (unsigned*)(ws + off); off += (size_t)N * 8;
  float*    den1 = ws + off; off += (size_t)N * 8;
  float*    agg1 = ws + off; off += (size_t)N * 512;
  off = (off + 7) & ~(size_t)7;                       // 32B align for v16h loads
  _Float16* bpack = (_Float16*)(ws + off); off += 16384;  // 32768 halves
  float*    h2   = ws + off; off += (size_t)N;
  float*    as2  = ws + off; off += (size_t)N;
  float*    ad2  = ws + off; off += (size_t)N;
  unsigned* m2   = (unsigned*)(ws + off); off += (size_t)N;
  float*    den2 = ws + off; off += (size_t)N;
  float*    agg2 = ws + off; off += (size_t)N;

  const int B = 256;

  // init accumulators / maxima every call (deterministic replay)
  k_fill_f32<<<blks((long)N * 512, B), B, 0, stream>>>(agg1, 0.f, (long)N * 512);
  k_fill_f32<<<blks((long)N * 8,   B), B, 0, stream>>>(den1, 0.f, (long)N * 8);
  k_fill_u32<<<blks((long)N * 8,   B), B, 0, stream>>>(m1, NEGINF_ORD, (long)N * 8);
  k_fill_f32<<<blks((long)N,       B), B, 0, stream>>>(den2, 0.f, (long)N);
  k_fill_f32<<<blks((long)N,       B), B, 0, stream>>>(agg2, 0.f, (long)N);
  k_fill_u32<<<blks((long)N,       B), B, 0, stream>>>(m2, NEGINF_ORD, (long)N);

  // pack W1 into WMMA B-fragment order (f16), then dense transform (WMMA)
  k_packB<<<blks(32768, B), B, 0, stream>>>(W1, bpack);
  {
    long tilesM = (N + 15) / 16;
    long waves  = tilesM * 8;                  // 8 groups of 4 column tiles
    k_gemm1_wmma<<<blks(waves * 32, B), B, 0, stream>>>(x, bpack, h1, N);
  }
  // attention coefficients
  k_att1<<<blks((long)N * 8, B), B, 0, stream>>>(h1, atS1, atD1, as1, ad1, N);

  // layer-1 edge softmax + aggregation (H=8, C=64)
  k_edge_max<<<blks(Etot * 8, B), B, 0, stream>>>(ei, E, N, 8, as1, ad1, m1);
  k_edge_sum<<<blks(Etot * 8, B), B, 0, stream>>>(ei, E, N, 8, as1, ad1, m1, den1);
  k_edge_agg<<<blks(Etot * 8, B), B, 0, stream>>>(ei, E, N, 8, 64, as1, ad1, m1, den1,
                                                  h1, agg1);

  // layer 2: ELU(agg1+b1) @ W2 -> h2, attention scalars
  k_gemv2<<<blks((long)N * 32, B), B, 0, stream>>>(agg1, b1, W2, atS2, atD2,
                                                   h2, as2, ad2, N);

  // layer-2 edge softmax + aggregation (H=1, C=1)
  k_edge_max<<<blks(Etot, B), B, 0, stream>>>(ei, E, N, 1, as2, ad2, m2);
  k_edge_sum<<<blks(Etot, B), B, 0, stream>>>(ei, E, N, 1, as2, ad2, m2, den2);
  k_edge_agg<<<blks(Etot, B), B, 0, stream>>>(ei, E, N, 1, 1, as2, ad2, m2, den2,
                                              h2, agg2);

  // final bias
  k_final<<<blks((long)N, B), B, 0, stream>>>(agg2, b2, (float*)d_out, N);
}
